// TreeMHA_70308614636050
// MI455X (gfx1250) — matrix operand order repaired and checked by reading, not compiled, not run on hardware
//
#include <hip/hip_runtime.h>

// ---------------------------------------------------------------------------
// Tree-Transformer MHA for MI455X (gfx1250), wave32, WMMA bf16 pipeline.
//   B=2, S=2048, D=1024, H=16, dh=64
// Pipeline: fp32->bf16 convert, 3x QKV projection GEMM (TDM-staged A tiles),
// flash-style attention (16 heads per block, LDS-shared mask/group_prob),
// output projection GEMM -> fp32.
// ---------------------------------------------------------------------------

typedef __bf16 bf16;
typedef __attribute__((ext_vector_type(16))) __bf16 v16bf;
typedef __attribute__((ext_vector_type(4)))  __bf16 bf16x4;
typedef __attribute__((ext_vector_type(8)))  float  v8f;
typedef __attribute__((ext_vector_type(4)))  float  f32x4;
typedef __attribute__((ext_vector_type(4)))  unsigned int u32x4;
typedef __attribute__((ext_vector_type(4)))  int    i32x4;
typedef __attribute__((ext_vector_type(8)))  int    i32x8;

#define SEQ    2048
#define DMODEL 1024
#define NH     16
#define DH     64
#define MTOT   4096   // B * SEQ rows

#if __has_builtin(__builtin_amdgcn_tensor_load_to_lds)
#define USE_TDM 1
#else
#define USE_TDM 0
#endif

union FragU { v16bf v; f32x4 q[2]; };

// A-matrix fragment (16x32, M x K), row-major source with row stride ld.
// ISA 7.12.2: lane h=0 holds row m, K = {0..7, 16..23}; h=1: K = {8..15, 24..31}.
__device__ __forceinline__ v16bf frag_ld_a(const bf16* base, int ld) {
  const int lane = threadIdx.x & 31;
  const int m = lane & 15, h = lane >> 4;
  const bf16* p = base + m * ld + 8 * h;
  FragU f;
  f.q[0] = *(const f32x4*)(p);        // K = 8h .. 8h+7
  f.q[1] = *(const f32x4*)(p + 16);   // K = 16+8h .. 16+8h+7
  return f.v;
}

// B-matrix fragment (32x16, K x N), stored as N x K (each "row" = one column n
// of B, K contiguous). Lane holds column n=lane&15, K = {8h..8h+7, 16+8h..16+8h+7}.
__device__ __forceinline__ v16bf frag_ld_b(const bf16* base, int ld) {
  const int lane = threadIdx.x & 31;
  const int n = lane & 15, h = lane >> 4;
  const bf16* p = base + n * ld + 8 * h;
  FragU f;
  f.q[0] = *(const f32x4*)(p);
  f.q[1] = *(const f32x4*)(p + 16);
  return f.v;
}

__device__ __forceinline__ v8f wmma_bf16(v16bf a, v16bf b, v8f c) {
  // (neg_a, A, neg_b, B, c_mod, C, reuse_a, reuse_b)
  return __builtin_amdgcn_wmma_f32_16x16x32_bf16(false, a, false, b,
                                                 (short)0, c, false, false);
}

// ---------------------------------------------------------------------------
// fp32 -> bf16 conversion (vectorized), n must be a multiple of 1024.
// ---------------------------------------------------------------------------
__global__ __launch_bounds__(256)
void cvt_kernel(const float* __restrict__ src, bf16* __restrict__ dst) {
  const size_t i = ((size_t)blockIdx.x * 256 + threadIdx.x) * 4;
  f32x4 f = *(const f32x4*)&src[i];
  bf16x4 o;
  o[0] = (bf16)f[0]; o[1] = (bf16)f[1]; o[2] = (bf16)f[2]; o[3] = (bf16)f[3];
  *(bf16x4*)&dst[i] = o;
}

// ---------------------------------------------------------------------------
// GEMM: D = A(4096 x 1024, bf16) * W(1024 x 1024, bf16) + bias(fp32).
// MODE 0: store bf16 per-head layout  [b, h, s, dh]   (Q / K)
// MODE 1: store bf16 transposed head  [b, h, dh, s]   (V)
// MODE 3: store fp32 row-major [row, col]             (final output)
// A tile staged into LDS by the Tensor Data Mover (TDM) with LDS padding
// (row = 16 DWORDs data + 8 DWORDs pad -> LDT=48 bf16), B tile staged
// transposed (N x K) by per-element loads.
// ---------------------------------------------------------------------------
template <int MODE>
__global__ __launch_bounds__(256)
void gemm_kernel(const bf16* __restrict__ A, const bf16* __restrict__ W,
                 const float* __restrict__ bias, void* __restrict__ Dptr) {
  constexpr int BM = 128, BN = 64, BK = 32, LDT = 48;
  constexpr int K = DMODEL, N = DMODEL;

  __shared__ __align__(16) bf16 sA[BM * LDT];   // M x K tile
  __shared__ __align__(16) bf16 sB[BN * LDT];   // N x K tile (B transposed)

  const int tid  = threadIdx.x;
  const int wave = tid >> 5;
  const int lane = tid & 31;
  const int wm = wave >> 1;       // 0..3  -> 32-row slice
  const int wn = wave & 1;        // 0..1  -> 32-col slice
  const int n0 = blockIdx.x * BN;
  const int m0 = blockIdx.y * BM;

  v8f acc[2][2] = {};

  for (int k0 = 0; k0 < K; k0 += BK) {
#if USE_TDM
    // ---- Tensor Data Mover: DMA the 128x32 bf16 A tile into padded LDS ----
    if (wave == 0) {
      const unsigned lds = (unsigned)(size_t)&sA[0];
      const unsigned long long ga =
          (unsigned long long)(size_t)&A[(size_t)m0 * K + k0];
      u32x4 g0 = {};
      g0[0] = 1u;                                        // count=1 valid D#
      g0[1] = lds;                                       // lds_addr
      g0[2] = (unsigned)(ga & 0xFFFFFFFFu);              // global_addr[31:0]
      g0[3] = (unsigned)((ga >> 32) & 0x1FFFFFFu) | (2u << 30);  // addr hi | type=2
      i32x8 g1 = {};
      // data_size=2B | pad_enable | pad_interval=16 DW | pad_amount=8 DW
      g1[0] = (1 << 16) | (1 << 20) | (3 << 22) | (7 << 25);
      g1[1] = (K & 0xFFFF) << 16;                        // tensor_dim0 lo
      g1[2] = ((K >> 16) & 0xFFFF) | ((MTOT & 0xFFFF) << 16);  // dim0 hi | dim1 lo
      g1[3] = ((MTOT >> 16) & 0xFFFF) | (BK << 16);      // dim1 hi | tile_dim0
      g1[4] = BM;                                        // tile_dim1 (tile_dim2=0)
      g1[5] = K;                                         // tensor_dim0_stride lo
      g1[6] = 0;
      g1[7] = 0;
      i32x4 z4 = {};
#if __clang_major__ >= 23
      i32x8 z8 = {};
      __builtin_amdgcn_tensor_load_to_lds(g0, g1, z4, z4, z8, 0);
#else
      __builtin_amdgcn_tensor_load_to_lds(g0, g1, z4, z4, 0);
#endif
      __builtin_amdgcn_s_wait_tensorcnt(0);
    }
#else
    // ---- fallback: vector-load staging of A tile ----
#pragma unroll
    for (int i = 0; i < 2; i++) {
      int idx = tid + i * 256;                 // 512 chunks of 8 bf16
      int row = idx >> 2;
      int c8 = (idx & 3) * 8;
      *(f32x4*)&sA[row * LDT + c8] =
          *(const f32x4*)&A[(size_t)(m0 + row) * K + k0 + c8];
    }
#endif
    // ---- stage B tile transposed: sB[n][k] = W[k0+k][n0+n] ----
#pragma unroll
    for (int i = 0; i < 8; i++) {
      int idx = tid + i * 256;                 // 2048 elements
      int kk = idx >> 6;
      int nn = idx & 63;
      sB[nn * LDT + kk] = W[(size_t)(k0 + kk) * N + n0 + nn];
    }
    if (k0 + BK < K)
      __builtin_prefetch(&W[(size_t)(k0 + BK + (tid >> 6)) * N + n0 + (tid & 63)], 0, 3);

    __syncthreads();

    const v16bf a0 = frag_ld_a(&sA[(wm * 32 + 0)  * LDT], LDT);
    const v16bf a1 = frag_ld_a(&sA[(wm * 32 + 16) * LDT], LDT);
    const v16bf b0 = frag_ld_b(&sB[(wn * 32 + 0)  * LDT], LDT);
    const v16bf b1 = frag_ld_b(&sB[(wn * 32 + 16) * LDT], LDT);
    acc[0][0] = wmma_bf16(a0, b0, acc[0][0]);
    acc[0][1] = wmma_bf16(a0, b1, acc[0][1]);
    acc[1][0] = wmma_bf16(a1, b0, acc[1][0]);
    acc[1][1] = wmma_bf16(a1, b1, acc[1][1]);

    __syncthreads();
  }

  // ---- epilogue ----
  const int lm = lane & 15, lh = lane >> 4;
#pragma unroll
  for (int i = 0; i < 2; i++)
#pragma unroll
    for (int j = 0; j < 2; j++)
#pragma unroll
      for (int r = 0; r < 8; r++) {
        int row = m0 + wm * 32 + i * 16 + r + 8 * lh;   // C layout: M = r + 8h
        int col = n0 + wn * 32 + j * 16 + lm;
        float val = acc[i][j][r] + bias[col];
        if (MODE == 0) {
          int b = row >> 11, s = row & (SEQ - 1);
          int hq = col >> 6, dd = col & (DH - 1);
          ((bf16*)Dptr)[((size_t)(b * NH + hq) * SEQ + s) * DH + dd] = (bf16)val;
        } else if (MODE == 1) {
          int b = row >> 11, s = row & (SEQ - 1);
          int hq = col >> 6, dd = col & (DH - 1);
          ((bf16*)Dptr)[((size_t)(b * NH + hq) * DH + dd) * SEQ + s] = (bf16)val;
        } else {
          ((float*)Dptr)[(size_t)row * DMODEL + col] = val;
        }
      }
}

// ---------------------------------------------------------------------------
// Attention: one block = (batch b, 16-query tile) x ALL 16 heads.
// 512 threads = 16 waves; wave w handles head w. group_prob / mask tiles are
// staged in LDS once and shared by all heads (kills the 16x re-read).
// Flash-style online softmax; post-softmax group_prob modulation folded into P.
// ---------------------------------------------------------------------------
__global__ __launch_bounds__(512)
void attn_kernel(const bf16* __restrict__ Qh, const bf16* __restrict__ Kh,
                 const bf16* __restrict__ Vt, const float* __restrict__ gp,
                 const int* __restrict__ mask, bf16* __restrict__ Oout) {
  __shared__ __align__(16) float s_gp[16 * 32];
  __shared__ __align__(16) float s_neg[16 * 32];
  __shared__ __align__(16) bf16  s_P[NH][16 * 32];   // per-wave P tile (16x32)

  const int tid  = threadIdx.x;
  const int head = tid >> 5;
  const int lane = tid & 31;
  const int ln = lane & 15;
  const int lh = lane >> 4;

  const int blk = blockIdx.x;          // B * (S/16) = 256 blocks
  const int b  = blk >> 7;
  const int q0 = (blk & 127) * 16;

  const bf16* Qbase = Qh + ((size_t)(b * NH + head) * SEQ + q0) * DH;
  const bf16* Kbase = Kh + (size_t)(b * NH + head) * SEQ * DH;
  const bf16* Vbase = Vt + (size_t)(b * NH + head) * DH * SEQ;
  const float* gpB = gp + (size_t)b * SEQ * SEQ + (size_t)q0 * SEQ;
  const int*  mkB = mask + (size_t)b * SEQ * SEQ + (size_t)q0 * SEQ;

  // Q tile 16x64 -> two A fragments (K halves), kept in registers for all kt.
  const v16bf qa0 = frag_ld_a(Qbase, DH);
  const v16bf qa1 = frag_ld_a(Qbase + 32, DH);

  v8f o0 = {}, o1 = {}, o2 = {}, o3 = {};    // O tile 16x64 (4 N-slices)
  float mrow[8], zrow[8];
#pragma unroll
  for (int r = 0; r < 8; r++) { mrow[r] = -1e30f; zrow[r] = 0.0f; }

  const float scale = 0.125f;   // 1/sqrt(dh)

  for (int kt = 0; kt < SEQ; kt += 32) {
    // ---- cooperative stage of group_prob / mask tile (16 x 32) ----
    {
      int rr = tid >> 5;                 // 0..15 query row
      int cc = tid & 31;                 // 0..31 key col
      size_t g = (size_t)rr * SEQ + kt + cc;
      s_gp[rr * 32 + cc]  = gpB[g];
      s_neg[rr * 32 + cc] = (1.0f - (float)mkB[g]) * -1e9f;
    }
    __syncthreads();

    // ---- logits = Q * K^T for 32 keys (two 16x16 C tiles) ----
    const bf16* Kt = Kbase + (size_t)kt * DH;
    v8f l0 = {}, l1 = {};
    l0 = wmma_bf16(qa0, frag_ld_b(Kt, DH), l0);
    l0 = wmma_bf16(qa1, frag_ld_b(Kt + 32, DH), l0);
    l1 = wmma_bf16(qa0, frag_ld_b(Kt + 16 * DH, DH), l1);
    l1 = wmma_bf16(qa1, frag_ld_b(Kt + 16 * DH + 32, DH), l1);

    // ---- online softmax (row reductions across 16-lane halves) ----
    float cf[8];
    bf16* P = &s_P[head][0];
#pragma unroll
    for (int r = 0; r < 8; r++) {
      int row = r + 8 * lh;
      float a0 = l0[r] * scale + s_neg[row * 32 + ln];
      float a1 = l1[r] * scale + s_neg[row * 32 + 16 + ln];
      float mx = fmaxf(a0, a1);
#pragma unroll
      for (int off = 1; off < 16; off <<= 1) mx = fmaxf(mx, __shfl_xor(mx, off, 32));
      float mnew = fmaxf(mrow[r], mx);
      float p0 = __expf(a0 - mnew);
      float p1 = __expf(a1 - mnew);
      float ps = p0 + p1;
#pragma unroll
      for (int off = 1; off < 16; off <<= 1) ps += __shfl_xor(ps, off, 32);
      float c = __expf(mrow[r] - mnew);
      zrow[r] = zrow[r] * c + ps;
      mrow[r] = mnew;
      cf[r] = c;
      // Tree-Transformer modulation: P = softmax-numerator * group_prob
      P[row * 32 + ln]      = (bf16)(p0 * s_gp[row * 32 + ln]);
      P[row * 32 + 16 + ln] = (bf16)(p1 * s_gp[row * 32 + 16 + ln]);
    }

    // rescale running O by exp(m_old - m_new)
#pragma unroll
    for (int r = 0; r < 8; r++) {
      o0[r] *= cf[r]; o1[r] *= cf[r]; o2[r] *= cf[r]; o3[r] *= cf[r];
    }

    __syncthreads();   // P (C-layout) visible for A-layout reload; gp tile free

    // ---- O += P(16x32) * V(32x64)  (V stored transposed [dh][s]) ----
    const v16bf pa = frag_ld_a(P, 32);
    o0 = wmma_bf16(pa, frag_ld_b(Vbase + 0 * 16 * SEQ + kt, SEQ), o0);
    o1 = wmma_bf16(pa, frag_ld_b(Vbase + 1 * 16 * SEQ + kt, SEQ), o1);
    o2 = wmma_bf16(pa, frag_ld_b(Vbase + 2 * 16 * SEQ + kt, SEQ), o2);
    o3 = wmma_bf16(pa, frag_ld_b(Vbase + 3 * 16 * SEQ + kt, SEQ), o3);
  }

  // ---- finalize: divide by Z, store merged-head bf16 [b, s, h*64+dh] ----
#pragma unroll
  for (int r = 0; r < 8; r++) {
    int row = r + 8 * lh;
    float iz = 1.0f / zrow[r];
    size_t base = ((size_t)b * SEQ + q0 + row) * DMODEL + head * DH;
    Oout[base + 0 * 16 + ln] = (bf16)(o0[r] * iz);
    Oout[base + 1 * 16 + ln] = (bf16)(o1[r] * iz);
    Oout[base + 2 * 16 + ln] = (bf16)(o2[r] * iz);
    Oout[base + 3 * 16 + ln] = (bf16)(o3[r] * iz);
  }
}

// ---------------------------------------------------------------------------
extern "C" void kernel_launch(void* const* d_in, const int* in_sizes, int n_in,
                              void* d_out, int out_size, void* d_ws, size_t ws_size,
                              hipStream_t stream) {
  const float* query  = (const float*)d_in[0];
  const float* key_in = (const float*)d_in[1];
  const float* value  = (const float*)d_in[2];
  const float* gp     = (const float*)d_in[3];
  const int*   mask   = (const int*)d_in[4];
  const float* wq  = (const float*)d_in[5];
  const float* wqb = (const float*)d_in[6];
  const float* wk  = (const float*)d_in[7];
  const float* wkb = (const float*)d_in[8];
  const float* wv  = (const float*)d_in[9];
  const float* wvb = (const float*)d_in[10];
  const float* wo  = (const float*)d_in[11];
  const float* wob = (const float*)d_in[12];
  float* out = (float*)d_out;

  // workspace layout (bf16 tensors)
  char* ws = (char*)d_ws;
  const size_t act = (size_t)MTOT * DMODEL * sizeof(bf16);   // 8 MB
  const size_t wsz = (size_t)DMODEL * DMODEL * sizeof(bf16); // 2 MB
  bf16* Qh = (bf16*)(ws + 0 * act);        // [b,h,s,dh]
  bf16* Kh = (bf16*)(ws + 1 * act);        // [b,h,s,dh]
  bf16* Vt = (bf16*)(ws + 2 * act);        // [b,h,dh,s]
  bf16* AO = (bf16*)(ws + 3 * act);        // [b,s,d] merged-head attention out
  bf16* Xq = (bf16*)(ws + 4 * act);        // bf16 copies of activations
  bf16* Xk = (bf16*)(ws + 5 * act);
  bf16* Xv = (bf16*)(ws + 6 * act);
  bf16* Wq = (bf16*)(ws + 7 * act + 0 * wsz);
  bf16* Wk = (bf16*)(ws + 7 * act + 1 * wsz);
  bf16* Wv = (bf16*)(ws + 7 * act + 2 * wsz);
  bf16* Wo = (bf16*)(ws + 7 * act + 3 * wsz);

  // ---- fp32 -> bf16 conversions ----
  const int actBlocks = (int)((size_t)MTOT * DMODEL / 1024);   // 4096
  const int wBlocks   = (int)((size_t)DMODEL * DMODEL / 1024); // 1024
  cvt_kernel<<<actBlocks, 256, 0, stream>>>(query,  Xq);
  cvt_kernel<<<actBlocks, 256, 0, stream>>>(key_in, Xk);
  cvt_kernel<<<actBlocks, 256, 0, stream>>>(value,  Xv);
  cvt_kernel<<<wBlocks, 256, 0, stream>>>(wq, Wq);
  cvt_kernel<<<wBlocks, 256, 0, stream>>>(wk, Wk);
  cvt_kernel<<<wBlocks, 256, 0, stream>>>(wv, Wv);
  cvt_kernel<<<wBlocks, 256, 0, stream>>>(wo, Wo);

  // ---- projections ----
  dim3 gg(DMODEL / 64, MTOT / 128), gb(256);
  gemm_kernel<0><<<gg, gb, 0, stream>>>(Xq, Wq, wqb, Qh);
  gemm_kernel<0><<<gg, gb, 0, stream>>>(Xk, Wk, wkb, Kh);
  gemm_kernel<1><<<gg, gb, 0, stream>>>(Xv, Wv, wvb, Vt);

  // ---- attention ----
  attn_kernel<<<dim3(2 * (SEQ / 16)), dim3(512), 0, stream>>>(Qh, Kh, Vt, gp, mask, AO);

  // ---- output projection ----
  gemm_kernel<3><<<gg, gb, 0, stream>>>(AO, Wo, wob, out);
}
